// BiSAN_bitree_V2_75488345195273
// MI455X (gfx1250) — compile-verified, matrix operand order checked by reading, not run on hardware
//
#include <hip/hip_runtime.h>
#include <math.h>

// ---------------------------------------------------------------------------
// BiSAN-bitree forward for MI455X (gfx1250).
// All dense layers -> one generic WMMA GEMM (bf16 in, f32 acc, 16x16x32),
// vectorized b128 fragment loads, branch-free K tail,
// sequential SSM scans -> register-resident VALU kernels,
// flips folded into direction-aware conv/scan, layout kept [B,S,E].
// ---------------------------------------------------------------------------

#define BB    32
#define SEQN  196
#define DIMN  256
#define NHEADS 8
#define M1    (BB*SEQN)       // 6272 rows  (mult of 64)
#define NJ    21
#define LCH   (SEQN*NJ)       // 4116
#define M2    (BB*LCH)        // 131712 rows (mult of 64)

typedef __attribute__((ext_vector_type(16))) __bf16   v16bf;
typedef __attribute__((ext_vector_type(8)))  float    v8f;
typedef __attribute__((ext_vector_type(8)))  unsigned v8u;

__device__ __constant__ int d_fchain[NJ] = {21,19,17,14,15,12,20,18,16,13,9,6,3,0,1,4,10,2,5,8,11};
__device__ __constant__ int d_bchain[NJ] = {11,8,5,2,10,4,1,0,3,6,9,13,16,18,20,12,15,14,17,19,21};

// ---------------- device helpers ----------------
// pack two fp32 into one dword of two bf16 (round-half-up)
__device__ inline unsigned pk_bf16(float f0, float f1) {
  unsigned u0 = __builtin_bit_cast(unsigned, f0);
  unsigned u1 = __builtin_bit_cast(unsigned, f1);
  return ((u1 + 0x8000u) & 0xffff0000u) | ((u0 + 0x8000u) >> 16);
}
__device__ inline float siluf(float x)     { return x / (1.f + __expf(-x)); }
__device__ inline float geluf(float x)     {
  float t = tanhf(0.7978845608028654f * (x + 0.044715f * x * x * x));
  return 0.5f * x * (1.f + t);
}
__device__ inline float softplusf(float x) { return (x > 20.f) ? x : log1pf(__expf(x)); }
__device__ inline float act_apply(float x, int act) {
  if (act == 1) return x > 0.f ? x : 0.f;
  if (act == 2) return siluf(x);
  if (act == 3) return geluf(x);
  return x;
}
// block = 256 threads = 8 waves (wave32)
__device__ inline float block_sum(float v, float* red) {
#pragma unroll
  for (int off = 16; off; off >>= 1) v += __shfl_xor(v, off, 32);
  int w = threadIdx.x >> 5, l = threadIdx.x & 31;
  if (l == 0) red[w] = v;
  __syncthreads();
  if (w == 0) {
    float s = (l < 8) ? red[l] : 0.f;
#pragma unroll
    for (int off = 4; off; off >>= 1) s += __shfl_xor(s, off, 32);
    if (l == 0) red[0] = s;
  }
  __syncthreads();
  float r = red[0];
  __syncthreads();
  return r;
}
__device__ inline float block_max(float v, float* red) {
#pragma unroll
  for (int off = 16; off; off >>= 1) v = fmaxf(v, __shfl_xor(v, off, 32));
  int w = threadIdx.x >> 5, l = threadIdx.x & 31;
  if (l == 0) red[w] = v;
  __syncthreads();
  if (w == 0) {
    float s = (l < 8) ? red[l] : -1e30f;
#pragma unroll
    for (int off = 4; off; off >>= 1) s = fmaxf(s, __shfl_xor(s, off, 32));
    if (l == 0) red[0] = s;
  }
  __syncthreads();
  float r = red[0];
  __syncthreads();
  return r;
}

// ---------------------------------------------------------------------------
// Generic GEMM: C[M,N] = act(A[M,K] @ W[N,K]^T + bias) (+=C if accum)
// fp32 inputs, bf16 WMMA 16x16x32, f32 accumulate.
// Block = 256 threads = 8 waves; block tile 64(M) x 32(N); one 16x16 per wave.
// A fragment (ISA 16-bit A 16x32 map): lane(0..15)=row; each lane's 16 elems
// are two contiguous 8-float runs at k0+half*8 and k0+16+half*8 -> b128 loads.
// B row pointer is clamped for n>=N (garbage cols masked at the store).
// ---------------------------------------------------------------------------
__global__ __launch_bounds__(256) void gemm_wmma(
    const float* __restrict__ A, const float* __restrict__ Wt,
    const float* __restrict__ bias, float* __restrict__ C,
    int M, int N, int K, int act, int accum)
{
  int lane = threadIdx.x & 31;
  int wave = threadIdx.x >> 5;
  int wm = wave >> 1;               // 0..3
  int wn = wave & 1;                // 0..1
  int m0 = blockIdx.x * 64 + wm * 16;
  int n0 = blockIdx.y * 32 + wn * 16;
  int half = lane >> 4;
  int l15  = lane & 15;
  int rowA = m0 + l15;              // M always a multiple of 64 -> in range
  int rowB = n0 + l15;
  int rowBc = (rowB < N) ? rowB : (N - 1);   // clamp; junk cols masked on store
  const float* Arow = A  + (size_t)rowA * K;
  const float* Brow = Wt + (size_t)rowBc * K;

  v8f acc = {};
  int Kmain = K & ~31;

  if ((K & 3) == 0) {
    // vector path: rows are 16B aligned (K multiple of 4, base 16B aligned)
    for (int k0 = 0; k0 < Kmain; k0 += 32) {
      const float4* ap = (const float4*)(Arow + k0 + half * 8);
      const float4* bp = (const float4*)(Brow + k0 + half * 8);
      float4 a0 = ap[0], a1 = ap[1], a4 = ap[4], a5 = ap[5];
      float4 b0 = bp[0], b1 = bp[1], b4 = bp[4], b5 = bp[5];
      v8u au, bu;
      au[0] = pk_bf16(a0.x, a0.y); au[1] = pk_bf16(a0.z, a0.w);
      au[2] = pk_bf16(a1.x, a1.y); au[3] = pk_bf16(a1.z, a1.w);
      au[4] = pk_bf16(a4.x, a4.y); au[5] = pk_bf16(a4.z, a4.w);
      au[6] = pk_bf16(a5.x, a5.y); au[7] = pk_bf16(a5.z, a5.w);
      bu[0] = pk_bf16(b0.x, b0.y); bu[1] = pk_bf16(b0.z, b0.w);
      bu[2] = pk_bf16(b1.x, b1.y); bu[3] = pk_bf16(b1.z, b1.w);
      bu[4] = pk_bf16(b4.x, b4.y); bu[5] = pk_bf16(b4.z, b4.w);
      bu[6] = pk_bf16(b5.x, b5.y); bu[7] = pk_bf16(b5.z, b5.w);
      acc = __builtin_amdgcn_wmma_f32_16x16x32_bf16(
          false, __builtin_bit_cast(v16bf, au),
          false, __builtin_bit_cast(v16bf, bu),
          (short)0, acc, false, false);
    }
  } else {
    // scalar path (only K=54 uses this): unguarded, addresses in range
    for (int k0 = 0; k0 < Kmain; k0 += 32) {
      v8u au, bu;
#pragma unroll
      for (int q = 0; q < 8; ++q) {
        int k = k0 + ((q < 4) ? 2 * q : 2 * q + 8) + half * 8;
        au[q] = pk_bf16(Arow[k], Arow[k + 1]);
        bu[q] = pk_bf16(Brow[k], Brow[k + 1]);
      }
      acc = __builtin_amdgcn_wmma_f32_16x16x32_bf16(
          false, __builtin_bit_cast(v16bf, au),
          false, __builtin_bit_cast(v16bf, bu),
          (short)0, acc, false, false);
    }
  }

  if (Kmain < K) {
    // one guarded tail iteration; clamp addresses + select-zero (no exec branch)
    v8u au, bu;
#pragma unroll
    for (int q = 0; q < 8; ++q) {
      int k = Kmain + ((q < 4) ? 2 * q : 2 * q + 8) + half * 8;
      int k0c = (k < K) ? k : (K - 1);
      int k1c = (k + 1 < K) ? (k + 1) : (K - 1);
      float fa0 = Arow[k0c], fa1 = Arow[k1c];
      float fb0 = Brow[k0c], fb1 = Brow[k1c];
      fa0 = (k     < K) ? fa0 : 0.f;
      fa1 = (k + 1 < K) ? fa1 : 0.f;
      fb0 = (k     < K) ? fb0 : 0.f;
      fb1 = (k + 1 < K) ? fb1 : 0.f;
      au[q] = pk_bf16(fa0, fa1);
      bu[q] = pk_bf16(fb0, fb1);
    }
    acc = __builtin_amdgcn_wmma_f32_16x16x32_bf16(
        false, __builtin_bit_cast(v16bf, au),
        false, __builtin_bit_cast(v16bf, bu),
        (short)0, acc, false, false);
  }

#pragma unroll
  for (int v = 0; v < 8; ++v) {
    int m = m0 + v + half * 8;
    int n = n0 + l15;
    if (n < N) {
      float r = acc[v];
      if (bias) r += bias[n];
      r = act_apply(r, act);
      size_t idx = (size_t)m * N + n;
      if (accum) r += C[idx];
      C[idx] = r;
    }
  }
}

// ---------------- layernorm over last dim 256 (optional residual) ----------
__global__ __launch_bounds__(256) void ln256(
    const float* __restrict__ x, const float* __restrict__ res,
    const float* __restrict__ w, const float* __restrict__ b,
    float* __restrict__ y)
{
  __shared__ float red[8];
  size_t row = blockIdx.x;
  int t = threadIdx.x;
  float v = x[row * 256 + t];
  if (res) v += res[row * 256 + t];
  float mean = block_sum(v, red) * (1.f / 256.f);
  float d = v - mean;
  float var = block_sum(d * d, red) * (1.f / 256.f);
  y[row * 256 + t] = d * rsqrtf(var + 1e-5f) * w[t] + b[t];
}

// ---------------- mamba gate + RMSNorm: out = rms(y*silu(z)) * nw ----------
__global__ __launch_bounds__(256) void gate_rms(
    const float* __restrict__ y, const float* __restrict__ zbase, int ldz,
    const float* __restrict__ nw, float* __restrict__ out, int D)
{
  __shared__ float red[8];
  size_t row = blockIdx.x;
  int t = threadIdx.x;
  float tv[2];
  int cnt = 0;
  float ss = 0.f;
  for (int i = t; i < D; i += 256) {
    float z = zbase[row * (size_t)ldz + i];
    float g = y[row * (size_t)D + i] * siluf(z);
    tv[cnt++] = g;
    ss += g * g;
  }
  float s = block_sum(ss, red);
  float inv = rsqrtf(s / (float)D + 1e-5f);
  cnt = 0;
  for (int i = t; i < D; i += 256)
    out[row * (size_t)D + i] = tv[cnt++] * inv * nw[i];
}

// ---------------- causal (or anti-causal) depthwise conv4 + silu -----------
__global__ void dwconv_silu(
    const float* __restrict__ zx, int in_off, int ldin,
    const float* __restrict__ cw, const float* __restrict__ cb,
    float* __restrict__ out, int Bb, int L, int C, int rev)
{
  int idx = blockIdx.x * blockDim.x + threadIdx.x;
  int total = Bb * L * C;
  if (idx >= total) return;
  int c = idx % C;
  int l = (idx / C) % L;
  int b = idx / (C * L);
  float s = cb[c];
#pragma unroll
  for (int j = 0; j < 4; ++j) {
    int ls = rev ? (l + 3 - j) : (l - 3 + j);
    int lc = (ls < 0) ? 0 : ((ls >= L) ? (L - 1) : ls);
    float xv = zx[((size_t)(b * L + lc)) * ldin + in_off + c];
    xv = (ls >= 0 && ls < L) ? xv : 0.f;          // select, not branch
    s += xv * cw[c * 4 + j];
  }
  out[idx] = siluf(s);
}

// ---------------- mamba2 selective scan (register state) -------------------
// one thread per (b, head, channel); dt recomputed per thread (cheap).
__global__ void mamba_scan(
    const float* __restrict__ zx, int ldin, int dt_off,
    const float* __restrict__ xbc, int d_inner,
    const float* __restrict__ A_log, const float* __restrict__ dt_bias,
    const float* __restrict__ Dp, float* __restrict__ y,
    int Bb, int L, int nh, int hd, int rev)
{
  int idx = blockIdx.x * blockDim.x + threadIdx.x;
  if (idx >= Bb * nh * hd) return;
  int p = idx % hd;
  int h = (idx / hd) % nh;
  int b = idx / (hd * nh);
  float A  = -__expf(A_log[h]);
  float db = dt_bias[h];
  float Dv = Dp[h];
  int conv_dim = d_inner + 32;
  float st[16];
#pragma unroll
  for (int n = 0; n < 16; ++n) st[n] = 0.f;
  for (int i = 0; i < L; ++i) {
    int l = rev ? (L - 1 - i) : i;
    size_t r = (size_t)(b * L + l);
    float dt  = softplusf(zx[r * ldin + dt_off + h] + db);
    float dec = __expf(dt * A);
    float xv  = xbc[r * conv_dim + h * hd + p];
    float dx  = dt * xv;
    const float4* B4 = (const float4*)(xbc + r * conv_dim + d_inner);
    float4 bq0 = B4[0], bq1 = B4[1], bq2 = B4[2], bq3 = B4[3];
    float4 cq0 = B4[4], cq1 = B4[5], cq2 = B4[6], cq3 = B4[7];
    float Bv[16] = {bq0.x,bq0.y,bq0.z,bq0.w, bq1.x,bq1.y,bq1.z,bq1.w,
                    bq2.x,bq2.y,bq2.z,bq2.w, bq3.x,bq3.y,bq3.z,bq3.w};
    float Cv[16] = {cq0.x,cq0.y,cq0.z,cq0.w, cq1.x,cq1.y,cq1.z,cq1.w,
                    cq2.x,cq2.y,cq2.z,cq2.w, cq3.x,cq3.y,cq3.z,cq3.w};
    float accv = 0.f;
#pragma unroll
    for (int n = 0; n < 16; ++n) {
      st[n] = st[n] * dec + dx * Bv[n];
      accv += st[n] * Cv[n];
    }
    y[r * d_inner + h * hd + p] = accv + xv * Dv;
  }
}

// ---------------- attention (per b,head,query-row block) -------------------
__global__ __launch_bounds__(256) void attn_kernel(
    const float* __restrict__ qkv, float* __restrict__ o)
{
  __shared__ float sc[SEQN];
  __shared__ float qv[32];
  __shared__ float red[8];
  int sq = blockIdx.x % SEQN;
  int h  = (blockIdx.x / SEQN) % NHEADS;
  int b  = blockIdx.x / (SEQN * NHEADS);
  int t = threadIdx.x;
  const int hd = 32;
  size_t qbase = ((size_t)(b * SEQN + sq)) * 768 + h * hd;
  if (t < hd) qv[t] = qkv[qbase + t];
  __syncthreads();
  float mymax = -1e30f;
  if (t < SEQN) {
    size_t kbase = ((size_t)(b * SEQN + t)) * 768 + 256 + h * hd;
    float s = 0.f;
#pragma unroll
    for (int d = 0; d < hd; ++d) s += qv[d] * qkv[kbase + d];
    s *= 0.17677669529663687f;   // 1/sqrt(32)
    sc[t] = s;
    mymax = s;
  }
  float m = block_max(mymax, red);
  float mysum = 0.f;
  if (t < SEQN) {
    float e = __expf(sc[t] - m);
    sc[t] = e;
    mysum = e;
  }
  float inv = 1.f / block_sum(mysum, red);
  if (t < hd) {
    float a = 0.f;
    for (int tk = 0; tk < SEQN; ++tk)
      a += sc[tk] * qkv[((size_t)(b * SEQN + tk)) * 768 + 512 + h * hd + t];
    o[((size_t)(b * SEQN + sq)) * 256 + h * hd + t] = a * inv;
  }
}

// ---------------- net_conv: out = silu(Wc @ x + b[t]) + resid --------------
__global__ void netconv_silu_add(
    const float* __restrict__ Wc, const float* __restrict__ cb,
    const float* __restrict__ xn, const float* __restrict__ resid,
    float* __restrict__ out)
{
  int idx = blockIdx.x * blockDim.x + threadIdx.x;
  if (idx >= M1 * DIMN) return;
  int d = idx & 255;
  int tpos = (idx >> 8) % SEQN;
  int b = idx / (256 * SEQN);
  float s = cb[tpos];
  const float* wr = Wc + (size_t)tpos * SEQN;
  for (int ss = 0; ss < SEQN; ++ss)
    s += wr[ss] * xn[((size_t)(b * SEQN + ss)) * 256 + d];
  out[idx] = siluf(s) + resid[idx];
}

// ---------------- FAN concat: [cos(pr) | sin(pr) | g] ----------------------
__global__ void fan_concat(const float* __restrict__ pr, const float* __restrict__ g,
                           float* __restrict__ cc, int M)
{
  int idx = blockIdx.x * blockDim.x + threadIdx.x;
  if (idx >= M * 512) return;
  int col = idx & 511;
  size_t row = idx >> 9;
  float v;
  if (col < 128)       v = __cosf(pr[row * 128 + col]);
  else if (col < 256)  v = __sinf(pr[row * 128 + col - 128]);
  else                 v = g[row * 256 + col - 256];
  cc[idx] = v;
}

__global__ void add_kernel(const float* a, const float* b, float* o, int n) {
  int i = blockIdx.x * blockDim.x + threadIdx.x;
  if (i < n) o[i] = a[i] + b[i];
}

// ---------------- tree chain gather / scatter-add --------------------------
__global__ void chain_gather(const float* __restrict__ xj, float* __restrict__ xg, int which) {
  int idx = blockIdx.x * blockDim.x + threadIdx.x;
  if (idx >= M2 * 64) return;
  int d = idx & 63;
  int r = idx >> 6;            // b*4116 + s*21 + i
  int i = r % NJ;
  int sb = r / NJ;             // b*196 + s
  int j = which ? d_bchain[i] : d_fchain[i];
  xg[idx] = xj[(size_t)sb * 1408 + j * 64 + d];
}
__global__ void chain_scatter(const float* __restrict__ og, float* __restrict__ full, int which) {
  int idx = blockIdx.x * blockDim.x + threadIdx.x;
  if (idx >= M2 * 64) return;
  int d = idx & 63;
  int r = idx >> 6;
  int i = r % NJ;
  int sb = r / NJ;
  int j = which ? d_bchain[i] : d_fchain[i];
  full[(size_t)sb * 1408 + j * 64 + d] += og[idx];
}

// ===========================================================================
// Host side
// ===========================================================================
struct MambaP { const float *A_log,*D,*conv_b,*conv_w,*dt_bias,*in_proj,*norm_w,*out_proj; };
struct FanP   { const float *g_b,*g_w,*out_b,*out_w,*p_w; };
struct TfP    { const float *b1,*b2,*in_proj_b,*in_proj_w,*ln1_b,*ln1_w,*ln2_b,*ln2_w,
                            *out_proj_b,*out_proj_w,*w1,*w2; };
struct BlockP {
  const float *emb1_b,*emb1_w,*emb2_b,*emb2_w;
  FanP fan, fan2;
  const float *lin0_b,*lin0_w;
  MambaP mb, mf;
  const float *net_conv_b,*net_conv_w,*net_ln_b,*net_ln_w,
              *norm0_b,*norm0_w,*norm1_b,*norm1_w;
  TfP tf;
};
struct WS {
  float *X,*XN,*X1,*X2,*H,*XM,*T0,*T1,*ZX,*XBC,*Y,*PR,*G,*CC,*XJ,*FULL,
        *XG,*ZXg,*XBCg,*Yg,*OG;
};

static void gemm(hipStream_t st, const float* A, const float* Wt, const float* bias,
                 float* C, int M, int N, int K, int act, int accum) {
  dim3 g(M / 64, (N + 31) / 32);
  gemm_wmma<<<g, 256, 0, st>>>(A, Wt, bias, C, M, N, K, act, accum);
}

static void run_mamba(hipStream_t st, const float* in, float* out, int accum,
                      const MambaP& p, int Bb, int L, int dm, int d_inner,
                      int hd, int rev, float* ZXb, float* XBCb, float* Yb) {
  int M = Bb * L;
  int nh = d_inner / hd;
  int conv_dim = d_inner + 32;
  int Nin = 2 * d_inner + 32 + nh;
  gemm(st, in, p.in_proj, nullptr, ZXb, M, Nin, dm, 0, 0);
  int tot = M * conv_dim;
  dwconv_silu<<<(tot + 255) / 256, 256, 0, st>>>(ZXb, d_inner, Nin, p.conv_w, p.conv_b,
                                                 XBCb, Bb, L, conv_dim, rev);
  int nthr = Bb * nh * hd;
  mamba_scan<<<(nthr + 255) / 256, 256, 0, st>>>(ZXb, Nin, d_inner + conv_dim, XBCb,
                                                 d_inner, p.A_log, p.dt_bias, p.D, Yb,
                                                 Bb, L, nh, hd, rev);
  gate_rms<<<M, 256, 0, st>>>(Yb, ZXb, Nin, p.norm_w, Yb, d_inner);
  gemm(st, Yb, p.out_proj, nullptr, out, M, dm, d_inner, 0, accum);
}

static void run_fan(hipStream_t st, const float* in, int din, const FanP& f,
                    float* out, int act, WS& w) {
  gemm(st, in, f.p_w, nullptr, w.PR, M1, 128, din, 0, 0);
  gemm(st, in, f.g_w, f.g_b,  w.G,  M1, 256, din, 3, 0);
  int tot = M1 * 512;
  fan_concat<<<(tot + 255) / 256, 256, 0, st>>>(w.PR, w.G, w.CC, M1);
  gemm(st, w.CC, f.out_w, f.out_b, out, M1, 256, 512, act, 0);
}

static void run_tail(hipStream_t st, const BlockP& bp, const float* x1, float* xout, WS& w) {
  int tot = M1 * DIMN;
  ln256<<<M1, 256, 0, st>>>(x1, nullptr, bp.net_ln_w, bp.net_ln_b, w.XN);
  netconv_silu_add<<<(tot + 255) / 256, 256, 0, st>>>(bp.net_conv_w, bp.net_conv_b,
                                                      w.XN, x1, w.X2);
  gemm(st, w.X2, bp.emb1_w, bp.emb1_b, w.H, M1, 256, 256, 0, 0);
  ln256<<<M1, 256, 0, st>>>(w.H, nullptr, bp.norm1_w, bp.norm1_b, w.H);
  // transformer layer
  gemm(st, w.H, bp.tf.in_proj_w, bp.tf.in_proj_b, w.T0, M1, 768, 256, 0, 0);
  attn_kernel<<<BB * NHEADS * SEQN, 256, 0, st>>>(w.T0, w.T1);
  gemm(st, w.T1, bp.tf.out_proj_w, bp.tf.out_proj_b, w.XN, M1, 256, 256, 0, 0);
  ln256<<<M1, 256, 0, st>>>(w.H, w.XN, bp.tf.ln1_w, bp.tf.ln1_b, w.H);
  gemm(st, w.H, bp.tf.w1, bp.tf.b1, w.T0, M1, 2048, 256, 1, 0);
  gemm(st, w.T0, bp.tf.w2, bp.tf.b2, w.T1, M1, 256, 2048, 0, 0);
  ln256<<<M1, 256, 0, st>>>(w.H, w.T1, bp.tf.ln2_w, bp.tf.ln2_b, w.H);
  // h = silu(fan(h))
  run_fan(st, w.H, 256, bp.fan, w.H, 2, w);
  gemm(st, w.H, bp.emb2_w, bp.emb2_b, w.XN, M1, 256, 256, 0, 0);
  add_kernel<<<(tot + 255) / 256, 256, 0, st>>>(w.XN, w.X2, xout, tot);
}

static void run_san(hipStream_t st, const BlockP& bp, WS& w) {
  int tot = M1 * DIMN;
  ln256<<<M1, 256, 0, st>>>(w.X, nullptr, bp.norm0_w, bp.norm0_b, w.XN);
  run_mamba(st, w.XN, w.XM, 0, bp.mf, BB, SEQN, 256, 512, 64, 0, w.ZX, w.XBC, w.Y);
  run_mamba(st, w.XN, w.XM, 1, bp.mb, BB, SEQN, 256, 512, 64, 1, w.ZX, w.XBC, w.Y);
  add_kernel<<<(tot + 255) / 256, 256, 0, st>>>(w.X, w.XM, w.X1, tot);
  run_tail(st, bp, w.X1, w.X, w);
}

static void run_tree(hipStream_t st, const BlockP& bp, WS& w) {
  int tot = M1 * DIMN;
  ln256<<<M1, 256, 0, st>>>(w.X, nullptr, bp.norm0_w, bp.norm0_b, w.XN);
  gemm(st, w.XN, bp.lin0_w, bp.lin0_b, w.XJ, M1, 1408, 256, 0, 0);
  hipMemsetAsync(w.FULL, 0, sizeof(float) * (size_t)M1 * 1408, st);
  for (int dir = 0; dir < 2; ++dir) {
    const MambaP& mp = dir ? bp.mb : bp.mf;
    int tg = M2 * 64;
    chain_gather<<<(tg + 255) / 256, 256, 0, st>>>(w.XJ, w.XG, dir);
    run_mamba(st, w.XG, w.OG, 0, mp, BB, LCH, 64, 128, 16, 0, w.ZXg, w.XBCg, w.Yg);
    chain_scatter<<<(tg + 255) / 256, 256, 0, st>>>(w.OG, w.FULL, dir);
  }
  run_fan(st, w.FULL, 1408, bp.fan2, w.XM, 0, w);
  add_kernel<<<(tot + 255) / 256, 256, 0, st>>>(w.X, w.XM, w.X1, tot);
  run_tail(st, bp, w.X1, w.X, w);
}

extern "C" void kernel_launch(void* const* d_in, const int* in_sizes, int n_in,
                              void* d_out, int out_size, void* d_ws, size_t ws_size,
                              hipStream_t stream) {
  (void)in_sizes; (void)n_in; (void)out_size; (void)ws_size;

  // ---- parameter parsing: jax pytree order (dict keys sorted, lists in order)
  int i = 0;
  const float* xin = (const float*)d_in[i++];           // input_tensor
  auto nx = [&]() { return (const float*)d_in[i++]; };
  const float* in_b  = nx();
  const float* in_w  = nx();
  const float* out_b = nx();
  const float* out_w = nx();
  auto readFan   = [&](FanP& f)   { f.g_b=nx(); f.g_w=nx(); f.out_b=nx(); f.out_w=nx(); f.p_w=nx(); };
  auto readMamba = [&](MambaP& m) { m.A_log=nx(); m.D=nx(); m.conv_b=nx(); m.conv_w=nx();
                                    m.dt_bias=nx(); m.in_proj=nx(); m.norm_w=nx(); m.out_proj=nx(); };
  auto readTf    = [&](TfP& t)    { t.b1=nx(); t.b2=nx(); t.in_proj_b=nx(); t.in_proj_w=nx();
                                    t.ln1_b=nx(); t.ln1_w=nx(); t.ln2_b=nx(); t.ln2_w=nx();
                                    t.out_proj_b=nx(); t.out_proj_w=nx(); t.w1=nx(); t.w2=nx(); };
  auto readCommonTail = [&](BlockP& b) {
    b.net_conv_b=nx(); b.net_conv_w=nx(); b.net_ln_b=nx(); b.net_ln_w=nx();
    b.norm0_b=nx(); b.norm0_w=nx(); b.norm1_b=nx(); b.norm1_w=nx();
    readTf(b.tf);
  };
  auto readSan  = [&](BlockP& b) {
    b.emb1_b=nx(); b.emb1_w=nx(); b.emb2_b=nx(); b.emb2_w=nx();
    readFan(b.fan);
    readMamba(b.mb); readMamba(b.mf);
    readCommonTail(b);
  };
  auto readTree = [&](BlockP& b) {
    b.emb1_b=nx(); b.emb1_w=nx(); b.emb2_b=nx(); b.emb2_w=nx();
    readFan(b.fan); readFan(b.fan2);
    b.lin0_b=nx(); b.lin0_w=nx();
    readMamba(b.mb); readMamba(b.mf);
    readCommonTail(b);
  };
  BlockP san0, san1, tree0, tree1;
  readSan(san0);  readSan(san1);
  readTree(tree0); readTree(tree1);

  // ---- workspace carve (floats)
  float* p = (float*)d_ws;
  auto take = [&](size_t n) { float* r = p; p += n; return r; };
  WS w;
  w.X    = take((size_t)M1 * DIMN);
  w.XN   = take((size_t)M1 * DIMN);
  w.X1   = take((size_t)M1 * DIMN);
  w.X2   = take((size_t)M1 * DIMN);
  w.H    = take((size_t)M1 * DIMN);
  w.XM   = take((size_t)M1 * DIMN);
  w.T0   = take((size_t)M1 * 2048);
  w.T1   = take((size_t)M1 * 2048);
  w.ZX   = take((size_t)M1 * 1064);
  w.XBC  = take((size_t)M1 * 544);
  w.Y    = take((size_t)M1 * 512);
  w.PR   = take((size_t)M1 * 128);
  w.G    = take((size_t)M1 * 256);
  w.CC   = take((size_t)M1 * 512);
  w.XJ   = take((size_t)M1 * 1408);
  w.FULL = take((size_t)M1 * 1408);
  w.XG   = take((size_t)M2 * 64);
  w.ZXg  = take((size_t)M2 * 296);
  w.XBCg = take((size_t)M2 * 160);
  w.Yg   = take((size_t)M2 * 128);
  w.OG   = take((size_t)M2 * 64);

  // ---- forward pass
  gemm(stream, xin, in_w, in_b, w.X, M1, 256, 54, 0, 0);
  run_san(stream, san0, w);
  run_san(stream, san1, w);
  run_tree(stream, tree0, w);
  run_tree(stream, tree1, w);
  gemm(stream, w.X, out_w, out_b, (float*)d_out, M1, 132, 256, 0, 0);
}
